// SinCosTanODELinear_84542136254844
// MI455X (gfx1250) — matrix operand order repaired and checked by reading, not compiled, not run on hardware
//
#include <hip/hip_runtime.h>
#include <cstdint>

// ---------------------------------------------------------------------------
// out = alpha * ( x@W + b*sin(x) + c*cos(x) + d*tan(x) )
// alpha = scalar RK4 factor (the reference ODE is linear in (y,z), y0=0).
//
// Roofline (MI455X): 132 MB min HBM traffic -> 5.7us floor @ 23.3 TB/s.
// f32 WMMA is 16x16x4 only (8x less K than bf16 16x16x32) -> use bf16x2
// split-precision: x@W ~= xh*Wh + xh*Wl + xl*Wh (3 bf16 WMMAs, f32 accum).
// Workspace: xh/xl 32+32 MB, WtH/WtL (transposed, N-major) 2+2 MB = 68 MB.
// ---------------------------------------------------------------------------

typedef __bf16 bf16_t;
typedef bf16_t v16bf __attribute__((ext_vector_type(16)));
typedef float  v8f   __attribute__((ext_vector_type(8)));
typedef unsigned int u32x4 __attribute__((ext_vector_type(4)));
typedef int i32x4 __attribute__((ext_vector_type(4)));
typedef unsigned short u16;

typedef __attribute__((address_space(1))) i32x4* gas_i32x4_p;  // global (AS1)
typedef __attribute__((address_space(3))) i32x4* las_i32x4_p;  // LDS (AS3)

#define HDIM   1024
#define BM     128
#define BN     128
#define KB     32
#define THREADS 256
#define LDK    40                      // padded LDS row stride (bf16 elems), 80B (16B aligned)
#define PLANE_BYTES (128 * LDK * 2)    // 10240
#define STAGE_BYTES (4 * PLANE_BYTES)  // Ah, Al, Bh, Bl
#define SMEM_BYTES  (2 * STAGE_BYTES)  // double buffered: 81920 (<320KB/WGP)

#if __has_builtin(__builtin_amdgcn_global_load_async_to_lds_b128)
#define HAVE_ASYNC 1
#else
#define HAVE_ASYNC 0
#endif

__device__ __forceinline__ void async_b128(const void* g, void* l) {
#if HAVE_ASYNC
  // CDNA5 GLOBAL_LOAD_ASYNC_TO_LDS_B128 (ASYNCcnt-tracked).
  // Param 0 is a typed int4 pointer in the device (AS1) address space per the
  // compiler diagnostic; param 1 is the LDS (AS3) destination.
  __builtin_amdgcn_global_load_async_to_lds_b128(
      (gas_i32x4_p)g, (las_i32x4_p)l, 0, 0);
#else
  *(u32x4*)l = *(const u32x4*)g;
#endif
}

__device__ __forceinline__ void wait_async_le8() {
#if HAVE_ASYNC
#if __has_builtin(__builtin_amdgcn_s_wait_asynccnt)
  __builtin_amdgcn_s_wait_asynccnt(8);
#else
  asm volatile("s_wait_asynccnt 8" ::: "memory");
#endif
#endif
}

// Split f32 into bf16 hi (truncated -> exact residual) + bf16 lo (RNE).
__device__ __forceinline__ void bf16_split(float x, u16& h, u16& l) {
  unsigned u  = __float_as_uint(x);
  unsigned hu = u & 0xffff0000u;
  float    r  = x - __uint_as_float(hu);
  unsigned ru = __float_as_uint(r);
  ru += 0x7fffu + ((ru >> 16) & 1u);
  h = (u16)(hu >> 16);
  l = (u16)(ru >> 16);
}

// ---------------------------- prepass kernels ------------------------------

__global__ void prep_x_kernel(const float* __restrict__ x,
                              u16* __restrict__ xh, u16* __restrict__ xl,
                              long n4) {
  long i = (long)blockIdx.x * blockDim.x + threadIdx.x;
  if (i >= n4) return;
  float4 v = reinterpret_cast<const float4*>(x)[i];
  u16 h0, h1, h2, h3, l0, l1, l2, l3;
  bf16_split(v.x, h0, l0); bf16_split(v.y, h1, l1);
  bf16_split(v.z, h2, l2); bf16_split(v.w, h3, l3);
  reinterpret_cast<ushort4*>(xh)[i] = make_ushort4(h0, h1, h2, h3);
  reinterpret_cast<ushort4*>(xl)[i] = make_ushort4(l0, l1, l2, l3);
}

// Wt[n][k] = W[k][n], split into bf16 hi/lo planes (LDS-tiled transpose).
__global__ void prep_w_kernel(const float* __restrict__ W,
                              u16* __restrict__ wth, u16* __restrict__ wtl) {
  __shared__ float tile[32][33];
  int n0 = blockIdx.x * 32, k0 = blockIdx.y * 32;
  for (int r = threadIdx.y; r < 32; r += 8)
    tile[r][threadIdx.x] = W[(size_t)(k0 + r) * HDIM + n0 + threadIdx.x];
  __syncthreads();
  for (int r = threadIdx.y; r < 32; r += 8) {
    float v = tile[threadIdx.x][r];   // = W[k0+tx][n0+r]
    u16 h, l; bf16_split(v, h, l);
    size_t o = (size_t)(n0 + r) * HDIM + k0 + threadIdx.x;
    wth[o] = h; wtl[o] = l;
  }
}

// -------------------------- fragment loads (LDS) ---------------------------
// A 16x32 bf16 (ISA 7.12.2): lane m=lane&15, half=lane>>4;
//   v0..3 = K 8h..8h+7 (contig, 16B), v4..7 = K 16+8h..23+8h (contig, 16B).
__device__ __forceinline__ v16bf load_fragA(const char* plane, int row, int half) {
  const char* p = plane + row * (LDK * 2) + half * 16;
  union { u32x4 q[2]; v16bf v; } u;
  u.q[0] = *(const u32x4*)p;
  u.q[1] = *(const u32x4*)(p + 32);
  return u.v;
}
// B 32x16 bf16: lane n=lane&15, half=lane>>4; K = 8h..8h+7, 16+8h..23+8h (same as A).
__device__ __forceinline__ v16bf load_fragB(const char* plane, int n, int half) {
  const char* p = plane + n * (LDK * 2) + half * 16;
  union { u32x4 q[2]; v16bf v; } u;
  u.q[0] = *(const u32x4*)p;
  u.q[1] = *(const u32x4*)(p + 32);
  return u.v;
}

// Stage one 128x32 bf16 plane: 512 16B chunks, 2 per thread.
__device__ __forceinline__ void copy_plane(const u16* __restrict__ g,
                                           char* lds, int tid) {
#pragma unroll
  for (int i = 0; i < 2; ++i) {
    int ch  = tid + i * THREADS;     // 0..511
    int row = ch >> 2;
    int cc  = (ch & 3) * 8;          // bf16 element offset (16B chunks)
    async_b128(g + (size_t)row * HDIM + cc, lds + row * (LDK * 2) + cc * 2);
  }
}

// ------------------------------ main kernel --------------------------------

__global__ __launch_bounds__(THREADS) void gemm_kernel(
    const u16* __restrict__ xh, const u16* __restrict__ xl,
    const u16* __restrict__ wth, const u16* __restrict__ wtl,
    const float* __restrict__ x, const float* __restrict__ b,
    const float* __restrict__ c, const float* __restrict__ d,
    float* __restrict__ out) {
  extern __shared__ char smem[];
  const int tid  = threadIdx.x;
  const int lane = tid & 31;
  const int l15  = lane & 15;
  const int half = lane >> 4;
  const int w    = tid >> 5;           // 8 waves: 2 (M) x 4 (N)
  const int mW   = (w >> 2) * 64;      // wave tile 64x32
  const int nW   = (w & 3) * 32;
  const int rowBase = blockIdx.x * BM;
  const int colBase = blockIdx.y * BN;

  const u16* gAh = xh  + (size_t)rowBase * HDIM;
  const u16* gAl = xl  + (size_t)rowBase * HDIM;
  const u16* gBh = wth + (size_t)colBase * HDIM;
  const u16* gBl = wtl + (size_t)colBase * HDIM;

  auto stage = [&](int buf, int k0) {
    char* s = smem + buf * STAGE_BYTES;
    copy_plane(gAh + k0, s + 0 * PLANE_BYTES, tid);
    copy_plane(gAl + k0, s + 1 * PLANE_BYTES, tid);
    copy_plane(gBh + k0, s + 2 * PLANE_BYTES, tid);
    copy_plane(gBl + k0, s + 3 * PLANE_BYTES, tid);
  };

  v8f acc[4][2] = {};
  const int NK = HDIM / KB;            // 32 K-steps
  stage(0, 0);

#pragma unroll 1
  for (int kt = 0; kt < NK; ++kt) {
    int knext = (kt + 1 < NK) ? (kt + 1) : (NK - 1);   // clamp: extra copy is benign
    stage((kt + 1) & 1, knext * KB);
    wait_async_le8();                  // prev stage's 8/wave done (in-order)
    __syncthreads();

    const char* s  = smem + (kt & 1) * STAGE_BYTES;
    const char* Ah = s + 0 * PLANE_BYTES;
    const char* Al = s + 1 * PLANE_BYTES;
    const char* Bh = s + 2 * PLANE_BYTES;
    const char* Bl = s + 3 * PLANE_BYTES;

    v16bf ah[4], al[4], bh[2], bl[2];
#pragma unroll
    for (int tm = 0; tm < 4; ++tm) {
      int r = mW + tm * 16 + l15;
      ah[tm] = load_fragA(Ah, r, half);
      al[tm] = load_fragA(Al, r, half);
    }
#pragma unroll
    for (int tn = 0; tn < 2; ++tn) {
      int n = nW + tn * 16 + l15;
      bh[tn] = load_fragB(Bh, n, half);
      bl[tn] = load_fragB(Bl, n, half);
    }
#pragma unroll
    for (int tm = 0; tm < 4; ++tm)
#pragma unroll
      for (int tn = 0; tn < 2; ++tn) {
        acc[tm][tn] = __builtin_amdgcn_wmma_f32_16x16x32_bf16(
            false, ah[tm], false, bh[tn], (short)0, acc[tm][tn], false, false);
        acc[tm][tn] = __builtin_amdgcn_wmma_f32_16x16x32_bf16(
            false, ah[tm], false, bl[tn], (short)0, acc[tm][tn], false, false);
        acc[tm][tn] = __builtin_amdgcn_wmma_f32_16x16x32_bf16(
            false, al[tm], false, bh[tn], (short)0, acc[tm][tn], false, false);
      }
    __syncthreads();
  }

  // alpha: scalar RK4 of y' = -y + t (z=1), ts = linspace(0,1,100)
  float a = 0.f;
  const float inv99 = 1.0f / 99.0f;
#pragma unroll 1
  for (int i = 0; i < 99; ++i) {
    float t  = i * inv99;
    float dt = (i + 1) * inv99 - t;
    float k1 = t - a;
    float th = t + 0.5f * dt;
    float k2 = th - (a + 0.5f * dt * k1);
    float k3 = th - (a + 0.5f * dt * k2);
    float k4 = (t + dt) - (a + dt * k3);
    a += dt * (1.0f / 6.0f) * (k1 + 2.0f * (k2 + k3) + k4);
  }
  const float alpha = a;

  // fused epilogue: + b*sin + c*cos + d*tan, scale by alpha
#pragma unroll
  for (int tn = 0; tn < 2; ++tn) {
    int col = colBase + nW + tn * 16 + l15;
    float bv = b[col], cv = c[col], dv = d[col];
#pragma unroll
    for (int tm = 0; tm < 4; ++tm) {
#pragma unroll
      for (int j = 0; j < 8; ++j) {
        int row = rowBase + mW + tm * 16 + half * 8 + j;   // C/D layout 7.12.2
        size_t o = (size_t)row * HDIM + col;
        float xv = x[o];
        float s  = __sinf(xv);
        float co = __cosf(xv);
        float e  = bv * s + cv * co + dv * __fdividef(s, co);
        out[o] = alpha * (acc[tm][tn][j] + e);
      }
    }
  }
}

// ------------------------------- launcher ----------------------------------

extern "C" void kernel_launch(void* const* d_in, const int* in_sizes, int n_in,
                              void* d_out, int out_size, void* d_ws, size_t ws_size,
                              hipStream_t stream) {
  const float* x = (const float*)d_in[0];
  const float* W = (const float*)d_in[1];
  const float* b = (const float*)d_in[2];
  const float* c = (const float*)d_in[3];
  const float* d = (const float*)d_in[4];
  float* out = (float*)d_out;
  const long H = HDIM;
  const long N = in_sizes[0] / H;      // 16384

  // workspace layout (requires ws_size >= (2N + 2H) * H * 2 bytes = 68 MB)
  u16* xh  = (u16*)d_ws;
  u16* xl  = xh + (size_t)N * H;
  u16* wth = xl + (size_t)N * H;
  u16* wtl = wth + (size_t)H * H;

  long n4 = N * H / 4;
  prep_x_kernel<<<(unsigned)((n4 + 255) / 256), 256, 0, stream>>>(x, xh, xl, n4);
  prep_w_kernel<<<dim3((unsigned)(H / 32), (unsigned)(H / 32)), dim3(32, 8), 0,
                  stream>>>(W, wth, wtl);

  (void)hipFuncSetAttribute(reinterpret_cast<const void*>(gemm_kernel),
                            hipFuncAttributeMaxDynamicSharedMemorySize,
                            SMEM_BYTES);
  gemm_kernel<<<dim3((unsigned)(N / BM), (unsigned)(H / BN)), THREADS,
                SMEM_BYTES, stream>>>(xh, xl, wth, wtl, x, b, c, d, out);
}
